// TemporalFeatureAggregator_70669391888942
// MI455X (gfx1250) — compile-verified, matrix operand order checked by reading, not run on hardware
//
#include <hip/hip_runtime.h>
#include <hip/hip_bf16.h>
#include <math.h>

typedef __attribute__((ext_vector_type(16))) _Float16 v16h;
typedef __attribute__((ext_vector_type(8)))  _Float16 v8h;
typedef __attribute__((ext_vector_type(8)))  float    v8f;

#define P_    16
#define N_    128
#define C_    256
#define S_    64
#define HID_  64
#define ROWT  264   // sXT row pitch in halves (>=256, mult of 8, odd 16-byte stride)
#define ROWTH 72    // sHT row pitch in halves (>=64,  mult of 8)

__device__ __forceinline__ float leakyf(float z)   { return z >= 0.f ? z : 0.01f * z; }
__device__ __forceinline__ float sigmoidf_(float z){ return 1.f / (1.f + __expf(-z)); }

// Two contiguous 8-half runs -> one 32-wide K fragment (lane-half split at kbase).
__device__ __forceinline__ v16h load_bfrag(const _Float16* rowp, int base) {
    v8h lo = *(const v8h*)(rowp + base);
    v8h hi = *(const v8h*)(rowp + base + 16);
    return __builtin_shufflevector(lo, hi, 0,1,2,3,4,5,6,7,8,9,10,11,12,13,14,15);
}

__global__ __launch_bounds__(256)
void TemporalFeatureAggregator_kernel(const float* __restrict__ x,
                                      const float* __restrict__ w1a,
                                      const float* __restrict__ w1b,
                                      const float* __restrict__ w3a,
                                      const float* __restrict__ w3b,
                                      float* __restrict__ out)
{
    // Transposed staging: row index = temporal position (with 1 halo row each
    // side), column index = channel. Makes WMMA B-fragments contiguous.
    __shared__ __align__(16) _Float16 sXT[66 * ROWT];      // sXT[s+1][c] = x[c][s]
    __shared__ __align__(16) _Float16 sHT[2][66 * ROWTH];  // sHT[b][s+1][h] = H[h][s]

    const int blk  = blockIdx.x;              // blk = p*128 + n
    const int p    = blk >> 7;
    const int tid  = threadIdx.x;
    const int wave = tid >> 5;
    const int lane = tid & 31;
    const int lm   = lane & 15;               // A: M-row / B: N-col / D: N
    const int lh   = lane >> 4;               // lane-half -> K sub-block
    const int kbase = lh << 3;

    const size_t xbase = (size_t)blk * (C_ * S_);
    const float* w1a_p = w1a + (size_t)p * (HID_ * C_ * 3);
    const float* w1b_p = w1b + (size_t)p * (C_ * HID_);
    const float* w3a_p = w3a + (size_t)p * (HID_ * C_ * 3);
    const float* w3b_p = w3b + (size_t)p * (C_ * HID_ * 3);

    // ---------------- Stage 1: stage x (f16, transposed) into LDS ------------
    // Coalesced float4 reads over s, scattered b16 writes (LDS transpose).
    for (int i = tid; i < (C_ * S_) / 4; i += 256) {
        int c  = i >> 4;            // 16 float4 per channel row
        int s0 = (i & 15) * 4;
        float4 v = *(const float4*)(x + xbase + (size_t)c * S_ + s0);
        sXT[(s0 + 1) * ROWT + c] = (_Float16)v.x;
        sXT[(s0 + 2) * ROWT + c] = (_Float16)v.y;
        sXT[(s0 + 3) * ROWT + c] = (_Float16)v.z;
        sXT[(s0 + 4) * ROWT + c] = (_Float16)v.w;
    }
    for (int i = tid; i < ROWT; i += 256) {   // zero halo rows of xT
        sXT[0 * ROWT + i]  = (_Float16)0.f;
        sXT[65 * ROWT + i] = (_Float16)0.f;
    }
    if (tid < ROWTH) {                        // zero halo rows of HT
        sHT[0][0 * ROWTH + tid]  = (_Float16)0.f;
        sHT[0][65 * ROWTH + tid] = (_Float16)0.f;
        sHT[1][0 * ROWTH + tid]  = (_Float16)0.f;
        sHT[1][65 * ROWTH + tid] = (_Float16)0.f;
    }
    __syncthreads();

    // ---------------- Stage 2: H[b] = leaky(conv3(x, w)) ---------------------
    // GEMM M=64(h) x N=64(s) x K=768 with K = (tap d)*256 + c.
    // wave -> (branch b = wave/4, h-tile mt = wave%4); 4 s-tiles per wave.
    {
        const int b  = wave >> 2;
        const int mt = wave & 3;
        const float* wp = b ? w3a_p : w1a_p;
        const int h = mt * 16 + lm;

        v8f acc[4] = {v8f{}, v8f{}, v8f{}, v8f{}};
        #pragma unroll 1
        for (int d = 0; d < 3; ++d) {
            #pragma unroll 1
            for (int cb = 0; cb < 8; ++cb) {
                v16h afrag;                    // strided (tap dim) -> scalar loads
                #pragma unroll
                for (int e = 0; e < 16; ++e) {
                    int kl = ((e & 8) << 1) | (e & 7) | kbase;   // 0..31
                    int c  = cb * 32 + kl;
                    afrag[e] = (_Float16)wp[((size_t)h * C_ + c) * 3 + d];
                }
                #pragma unroll
                for (int nt = 0; nt < 4; ++nt) {
                    // B[(d,c), s] = x[c, s+d-1] = sXT[s+d][c]  (contiguous in c)
                    const _Float16* rowp = &sXT[(nt * 16 + lm + d) * ROWT];
                    v16h bfrag = load_bfrag(rowp, cb * 32 + kbase);
                    acc[nt] = __builtin_amdgcn_wmma_f32_16x16x32_f16(
                        false, afrag, false, bfrag, (short)0, acc[nt], false, false);
                }
            }
        }
        // Epilogue: leaky -> f16, one b128 store per tile (8 consecutive h).
        #pragma unroll
        for (int nt = 0; nt < 4; ++nt) {
            int s = nt * 16 + lm;
            v8h hv;
            #pragma unroll
            for (int r = 0; r < 8; ++r)
                hv[r] = (_Float16)leakyf(acc[nt][r]);   // h = mt*16 + 8*lh + r
            *(v8h*)(&sHT[b][(s + 1) * ROWTH + mt * 16 + 8 * lh]) = hv;
        }
    }
    __syncthreads();

    // ------- Stage 3: logits GEMMs + pooling gates + temporal max (fused) ----
    const float* xg   = x + xbase;
    float*       outp = out + (size_t)blk * C_;

    #pragma unroll 1
    for (int mi = 0; mi < 2; ++mi) {
        const int mt = wave * 2 + mi;          // channel tile 0..15
        const int cc = mt * 16 + lm;           // A-row channel for this lane
        float cmax[8];
        #pragma unroll
        for (int r = 0; r < 8; ++r) cmax[r] = -INFINITY;

        #pragma unroll 1
        for (int nt = 0; nt < 4; ++nt) {
            v8f a1 = {}, a3 = {};

            // logits1[c,s] = sum_h w1b[c,h] * H1[h,s]   (K = 64)
            #pragma unroll 1
            for (int kk = 0; kk < 2; ++kk) {
                // A is contiguous in k: two float4 pairs + cvt.
                const float4* f = (const float4*)(w1b_p + cc * HID_ + kk * 32 + kbase);
                float4 a0 = f[0], a1v = f[1], a2 = f[4], a3v = f[5];
                v16h af;
                af[0]=(_Float16)a0.x;  af[1]=(_Float16)a0.y;
                af[2]=(_Float16)a0.z;  af[3]=(_Float16)a0.w;
                af[4]=(_Float16)a1v.x; af[5]=(_Float16)a1v.y;
                af[6]=(_Float16)a1v.z; af[7]=(_Float16)a1v.w;
                af[8]=(_Float16)a2.x;  af[9]=(_Float16)a2.y;
                af[10]=(_Float16)a2.z; af[11]=(_Float16)a2.w;
                af[12]=(_Float16)a3v.x; af[13]=(_Float16)a3v.y;
                af[14]=(_Float16)a3v.z; af[15]=(_Float16)a3v.w;
                // B: H1[h][s] = sHT[0][s+1][h]  (contiguous in h)
                v16h bf = load_bfrag(&sHT[0][(nt * 16 + lm + 1) * ROWTH],
                                     kk * 32 + kbase);
                a1 = __builtin_amdgcn_wmma_f32_16x16x32_f16(
                    false, af, false, bf, (short)0, a1, false, false);
            }

            // logits3[c,s] = sum_{h,d} w3b[c,h,d] * H3[h,s+d-1]   (K = 192)
            #pragma unroll 1
            for (int kk = 0; kk < 6; ++kk) {
                int d  = kk >> 1;
                int hb = (kk & 1) * 32;
                v16h af;                       // stride-3 (tap dim) -> scalar loads
                #pragma unroll
                for (int e = 0; e < 16; ++e) {
                    int kl = ((e & 8) << 1) | (e & 7) | kbase;
                    int hh = hb + kl;
                    af[e] = (_Float16)w3b_p[((size_t)cc * HID_ + hh) * 3 + d];
                }
                // B: H3[h][s+d-1] = sHT[1][s+d][h]  (contiguous in h)
                v16h bf = load_bfrag(&sHT[1][(nt * 16 + lm + d) * ROWTH],
                                     hb + kbase);
                a3 = __builtin_amdgcn_wmma_f32_16x16x32_f16(
                    false, af, false, bf, (short)0, a3, false, false);
            }

            // Epilogue: pooled gates (f32 x from L2) + max over s within tile.
            #pragma unroll
            for (int r = 0; r < 8; ++r) {
                int c = mt * 16 + r + 8 * lh;
                int s = nt * 16 + lm;
                const float* xr = xg + c * S_;
                float x0  = xr[s];
                float xm1 = (s >= 1)  ? xr[s - 1] : 0.f;
                float xp1 = (s <= 62) ? xr[s + 1] : 0.f;
                float xm2 = (s >= 2)  ? xr[s - 2] : 0.f;
                float xp2 = (s <= 61) ? xr[s + 2] : 0.f;
                float avg3 = (xm1 + x0 + xp1) * (1.f / 3.f);   // count_include_pad
                float avg5 = (xm2 + xm1 + x0 + xp1 + xp2) * 0.2f;
                float mx3 = x0;                                 // max: valid taps only
                if (s >= 1)  mx3 = fmaxf(mx3, xm1);
                if (s <= 62) mx3 = fmaxf(mx3, xp1);
                float mx5 = mx3;
                if (s >= 2)  mx5 = fmaxf(mx5, xm2);
                if (s <= 61) mx5 = fmaxf(mx5, xp2);

                float feat = (avg3 + mx3) * sigmoidf_(a1[r])
                           + (avg5 + mx5) * sigmoidf_(a3[r]);

                // temporal max over the 16 s-lanes of this half-wave
                #pragma unroll
                for (int off = 8; off >= 1; off >>= 1)
                    feat = fmaxf(feat, __shfl_xor(feat, off, 32));
                cmax[r] = fmaxf(cmax[r], feat);
            }
        }
        if (lm == 0) {
            #pragma unroll
            for (int r = 0; r < 8; ++r)
                outp[mt * 16 + 8 * lh + r] = cmax[r];
        }
    }
}

extern "C" void kernel_launch(void* const* d_in, const int* in_sizes, int n_in,
                              void* d_out, int out_size, void* d_ws, size_t ws_size,
                              hipStream_t stream) {
    const float* x   = (const float*)d_in[0];
    const float* w1a = (const float*)d_in[1];
    const float* w1b = (const float*)d_in[2];
    const float* w3a = (const float*)d_in[3];
    const float* w3b = (const float*)d_in[4];
    float* out = (float*)d_out;

    dim3 grid(P_ * N_);   // one workgroup per (part, sample)
    dim3 block(256);      // 8 wave32 waves
    TemporalFeatureAggregator_kernel<<<grid, block, 0, stream>>>(
        x, w1a, w1b, w3a, w3b, out);
}
